// FlashAttention_10668698764046
// MI455X (gfx1250) — matrix operand order repaired and checked by reading, not compiled, not run on hardware
//
#include <hip/hip_runtime.h>

// Problem constants (from reference): B=4, S=2048, D=1024, H=16, HD=64
constexpr int CB = 4;
constexpr int CS = 2048;
constexpr int CD = 1024;
constexpr int CH = 16;
constexpr int CHD = 64;
// SCALE = HD^-0.5 = 0.125

typedef __attribute__((ext_vector_type(16))) __bf16 v16bf;
typedef __attribute__((ext_vector_type(8)))  float  v8f;
typedef __attribute__((ext_vector_type(4)))  unsigned int v4u;

// GCC-style vectors matching the async-builtin parameter types exactly
typedef int gvec4 __attribute__((vector_size(16)));
typedef int gvec2 __attribute__((vector_size(8)));
#define GAS __attribute__((address_space(1)))
#define LAS __attribute__((address_space(3)))

union FragU { v16bf v; v4u q[2]; };

#if __has_builtin(__builtin_amdgcn_global_load_async_to_lds_b128) && \
    __has_builtin(__builtin_amdgcn_global_load_async_to_lds_b64)
#define HAS_ASYNC 1
#else
#define HAS_ASYNC 0
#endif

// 16-byte global -> LDS copy (async on CDNA5: tracked by ASYNCcnt)
__device__ __forceinline__ void cp16(__bf16* lds_dst, const __bf16* gsrc) {
#if HAS_ASYNC
  __builtin_amdgcn_global_load_async_to_lds_b128(
      (GAS gvec4*)gsrc, (LAS gvec4*)lds_dst, 0, 0);
#else
  *reinterpret_cast<v4u*>(lds_dst) = *reinterpret_cast<const v4u*>(gsrc);
#endif
}
// 8-byte global -> LDS copy
__device__ __forceinline__ void cp8(__bf16* lds_dst, const __bf16* gsrc) {
#if HAS_ASYNC
  __builtin_amdgcn_global_load_async_to_lds_b64(
      (GAS gvec2*)gsrc, (LAS gvec2*)lds_dst, 0, 0);
#else
  *reinterpret_cast<unsigned long long*>(lds_dst) =
      *reinterpret_cast<const unsigned long long*>(gsrc);
#endif
}
template <int N>
__device__ __forceinline__ void cp_wait() {
#if HAS_ASYNC
#if __has_builtin(__builtin_amdgcn_s_wait_asynccnt)
  __builtin_amdgcn_s_wait_asynccnt(N);
#else
  asm volatile("s_wait_asynccnt %0" :: "i"(N) : "memory");
#endif
#endif
}

__device__ __forceinline__ __bf16 f2bf(float f) {
  unsigned u = __builtin_bit_cast(unsigned, f);
  u += 0x7FFFu + ((u >> 16) & 1u);          // round-to-nearest-even
  unsigned short h = (unsigned short)(u >> 16);
  return __builtin_bit_cast(__bf16, h);
}

// A-matrix fragment (16x32 bf16, MxK), per ISA table:
// lane<16: M=lane, K={0..7,16..23}; lane>=16: same M, K={8..15,24..31}
__device__ __forceinline__ v16bf load_fragA(const __bf16* base, int ld, int lane) {
  const int m  = lane & 15;
  const int kb = (lane >> 4) * 8;
  const __bf16* r = base + m * ld + kb;
  FragU f;
  f.q[0] = *reinterpret_cast<const v4u*>(r);
  f.q[1] = *reinterpret_cast<const v4u*>(r + 16);
  return f.v;
}

// B-matrix fragment (32x16 bf16, KxN) from B^T stored row-major [N][K]
__device__ __forceinline__ v16bf load_fragB(const __bf16* base, int ld, int lane) {
  const int n  = lane & 15;
  const int kb = (lane >> 4) * 8;
  const __bf16* r = base + n * ld + kb;
  FragU f;
  f.q[0] = *reinterpret_cast<const v4u*>(r);
  f.q[1] = *reinterpret_cast<const v4u*>(r + 16);
  return f.v;
}

__device__ __forceinline__ v8f wmma_bf16(v16bf a, v16bf b, v8f c) {
  return __builtin_amdgcn_wmma_f32_16x16x32_bf16(
      false, a, false, b, (short)0, c, false, false);
}

// ---------------------------------------------------------------------------
// Kernel 0: f32 -> bf16 conversion. X copied; weights transposed to [N][K].
// ---------------------------------------------------------------------------
__global__ void __launch_bounds__(256)
convert_kernel(const float* __restrict__ X, const float* __restrict__ Wqkv,
               const float* __restrict__ Wout, __bf16* __restrict__ Xbf,
               __bf16* __restrict__ Wqkvt, __bf16* __restrict__ Woutt) {
  const int i = blockIdx.x * blockDim.x + threadIdx.x;   // grid == CB*CS*CD
  Xbf[i] = f2bf(X[i]);
  if (i < 3 * CD * CD) {
    const int n = i >> 10, d = i & 1023;
    Wqkvt[i] = f2bf(Wqkv[d * (3 * CD) + n]);
  }
  if (i < CD * CD) {
    const int n = i >> 10, d = i & 1023;
    Woutt[i] = f2bf(Wout[d * CD + n]);
  }
}

// ---------------------------------------------------------------------------
// Shared GEMM shape: 128x128 block tile, 8 waves (2x4), wave 64x32.
// Double-buffered async LDS staging of A/B tiles (4 async ops/thread/stage).
// ---------------------------------------------------------------------------
constexpr int LDT = 40;  // padded LDS row stride for 32-elem rows (80B, 16B-aligned)

// ---------------------------------------------------------------------------
// Kernel 1: QKV projection GEMM (M=8192, N=3072, K=1024).
// Scatters into Q[b,h,s,hd] (pre-scaled by 0.125), K[b,h,s,hd], V^T[b,h,hd,s].
// ---------------------------------------------------------------------------
__global__ void __launch_bounds__(256)
qkv_gemm_kernel(const __bf16* __restrict__ A, const __bf16* __restrict__ Bt,
                __bf16* __restrict__ Qw, __bf16* __restrict__ Kw,
                __bf16* __restrict__ VTw) {
  __shared__ __align__(16) __bf16 Abuf[2][128 * LDT];
  __shared__ __align__(16) __bf16 Bbuf[2][128 * LDT];
  const int tid = threadIdx.x;
  const int lane = tid & 31, wave = tid >> 5;
  const int wm = wave >> 2, wn = wave & 3;
  const int m0 = blockIdx.y * 128, n0 = blockIdx.x * 128;
  const __bf16* Ag = A + m0 * CD;
  const __bf16* Bg = Bt + n0 * CD;

  auto stage = [&](int buf, int k0) {
    for (int c = tid; c < 512; c += 256) {          // 2 iterations
      const int row = c >> 2, col = (c & 3) * 8;    // 16B chunks of a 128x32 tile
      cp16(&Abuf[buf][row * LDT + col], Ag + row * CD + k0 + col);
      cp16(&Bbuf[buf][row * LDT + col], Bg + row * CD + k0 + col);
    }
  };

  v8f acc[4][2] = {};
  stage(0, 0);
  for (int k0 = 0; k0 < CD; k0 += 32) {
    const int cur = (k0 >> 5) & 1;
    if (k0 + 32 < CD) { stage(cur ^ 1, k0 + 32); cp_wait<4>(); }
    else              { cp_wait<0>(); }
    __syncthreads();
    v16bf a[4], bb[2];
#pragma unroll
    for (int i = 0; i < 4; ++i)
      a[i] = load_fragA(&Abuf[cur][(wm * 64 + i * 16) * LDT], LDT, lane);
#pragma unroll
    for (int j = 0; j < 2; ++j)
      bb[j] = load_fragB(&Bbuf[cur][(wn * 32 + j * 16) * LDT], LDT, lane);
#pragma unroll
    for (int i = 0; i < 4; ++i)
#pragma unroll
      for (int j = 0; j < 2; ++j)
        acc[i][j] = wmma_bf16(a[i], bb[j], acc[i][j]);
    __syncthreads();
  }

  // Epilogue: `which` is uniform per block (128-wide N tiles never straddle 1024)
  const int which = n0 >> 10;
  const int half = lane >> 4, ln = lane & 15;
  const int mbase = m0 + wm * 64;
  const int dbase = (n0 & 1023) + wn * 32;
#pragma unroll
  for (int i = 0; i < 4; ++i)
#pragma unroll
    for (int r = 0; r < 8; ++r) {
      const int row = mbase + i * 16 + r + 8 * half;
      const int b = row >> 11, s = row & (CS - 1);
#pragma unroll
      for (int j = 0; j < 2; ++j) {
        const int d = dbase + j * 16 + ln;
        const int h = d >> 6, hd = d & 63;
        const int bh = b * CH + h;
        const float val = acc[i][j][r];
        if (which == 0)      Qw[(bh * CS + s) * CHD + hd] = f2bf(val * 0.125f);
        else if (which == 1) Kw[(bh * CS + s) * CHD + hd] = f2bf(val);
        else                 VTw[(bh * CHD + hd) * CS + s] = f2bf(val);
      }
    }
}

// ---------------------------------------------------------------------------
// Kernel 2: flash attention. Block = 128 q-rows of one (b,h); 8 waves, each
// owns 16 q-rows. K/V tiles (shared by all 8 waves) staged via async->LDS,
// double-buffered. Online softmax; 8 WMMAs per 32-key step.
// ---------------------------------------------------------------------------
constexpr int LDK = 72;  // 64-elem K rows + pad (144B, 16B-aligned)
constexpr int LDV = 40;  // 32-elem V^T rows + pad
constexpr int LDP = 40;  // 32-elem P rows + pad

__global__ void __launch_bounds__(256)
attn_kernel(const __bf16* __restrict__ Qw, const __bf16* __restrict__ Kw,
            const __bf16* __restrict__ VTw, __bf16* __restrict__ Ctx) {
  const int tid = threadIdx.x;
  const int lane = tid & 31, wave = tid >> 5;
  const int half = lane >> 4, ln = lane & 15;
  const int bh = blockIdx.y;
  const int b = bh >> 4, h = bh & 15;
  const int q0 = blockIdx.x * 128 + wave * 16;

  const __bf16* Qb = Qw + (size_t)bh * CS * CHD;
  const __bf16* Kb = Kw + (size_t)bh * CS * CHD;
  const __bf16* Vt = VTw + (size_t)bh * CHD * CS;

  __shared__ __align__(16) __bf16 Ktile[2][32 * LDK];   // [s_k][hd]
  __shared__ __align__(16) __bf16 Vtile[2][64 * LDV];   // [hd][s_k]
  __shared__ __align__(16) __bf16 Plds[8][16 * LDP];

  // Every thread issues exactly 2 async ops per stage (uniform ASYNCcnt):
  // one 16B chunk of the 32x64 K tile, one 8B chunk of the 64x32 V^T tile.
  auto stage = [&](int buf, int kb0) {
    {
      const int row = tid >> 2, col = (tid & 3) * 8;    // 256 x 16B chunks
      cp16(&Ktile[buf][row * LDK + col], Kb + (kb0 + row) * CHD + col);
    }
    {
      const int row = tid >> 2, col = (tid & 3) * 4;    // 256 x 8B chunks
      cp8(&Vtile[buf][row * LDV + col], Vt + row * CS + kb0 + col);
    }
  };

  // Q fragments (16 rows x 64 hd), resident in VGPRs for the whole pass
  const v16bf qf0 = load_fragA(Qb + q0 * CHD + 0, CHD, lane);
  const v16bf qf1 = load_fragA(Qb + q0 * CHD + 32, CHD, lane);

  v8f O[4] = {};
  float mrow[8], lrow[8];
#pragma unroll
  for (int r = 0; r < 8; ++r) { mrow[r] = -3.0e38f; lrow[r] = 0.0f; }

  stage(0, 0);
  for (int kb0 = 0; kb0 < CS; kb0 += 32) {
    const int cur = (kb0 >> 5) & 1;
    if (kb0 + 32 < CS) { stage(cur ^ 1, kb0 + 32); cp_wait<2>(); }
    else               { cp_wait<0>(); }
    __syncthreads();                      // publish cur K/V tiles

    // scores: 16q x 32k, contraction over hd=64 (2 WMMAs per 16x16 tile)
    v8f sc[2];
#pragma unroll
    for (int t = 0; t < 2; ++t) {
      v8f c = {};
      const v16bf kf0 = load_fragB(&Ktile[cur][(16 * t) * LDK + 0], LDK, lane);
      const v16bf kf1 = load_fragB(&Ktile[cur][(16 * t) * LDK + 32], LDK, lane);
      c = wmma_bf16(qf0, kf0, c);
      c = wmma_bf16(qf1, kf1, c);
      sc[t] = c;
    }

    // online softmax; C layout: (q=r+8*half, k=16*t+ln) lives in lane ln+16*half
    float alpha[8];
#pragma unroll
    for (int r = 0; r < 8; ++r) {
      const float v0 = sc[0][r], v1 = sc[1][r];
      float mx = fmaxf(v0, v1);
#pragma unroll
      for (int d = 1; d < 16; d <<= 1) mx = fmaxf(mx, __shfl_xor(mx, d, 32));
      const float mnew = fmaxf(mrow[r], mx);
      const float a = __expf(mrow[r] - mnew);
      const float p0 = __expf(v0 - mnew);
      const float p1 = __expf(v1 - mnew);
      float ps = p0 + p1;
#pragma unroll
      for (int d = 1; d < 16; d <<= 1) ps += __shfl_xor(ps, d, 32);
      lrow[r] = lrow[r] * a + ps;
      mrow[r] = mnew;
      alpha[r] = a;
      const int prow = r + 8 * half;
      Plds[wave][prow * LDP + ln]      = f2bf(p0);
      Plds[wave][prow * LDP + 16 + ln] = f2bf(p1);
    }
    __builtin_amdgcn_wave_barrier();      // P region is wave-private; LDS is
                                          // in-order per wave — just fence codegen
    const v16bf pf = load_fragA(&Plds[wave][0], LDP, lane);
#pragma unroll
    for (int j = 0; j < 4; ++j) {
      v8f o = O[j];
#pragma unroll
      for (int r = 0; r < 8; ++r) o[r] *= alpha[r];     // rescale running output
      const v16bf vf = load_fragB(&Vtile[cur][(16 * j) * LDV], LDV, lane);
      O[j] = wmma_bf16(pf, vf, o);
    }
    __syncthreads();                      // all reads of cur done before restage
  }

#pragma unroll
  for (int r = 0; r < 8; ++r) lrow[r] = 1.0f / lrow[r];
#pragma unroll
  for (int j = 0; j < 4; ++j)
#pragma unroll
    for (int r = 0; r < 8; ++r) {
      const int s = q0 + r + 8 * half;
      const int col = h * CHD + 16 * j + ln;
      Ctx[(b * CS + s) * CD + col] = f2bf(O[j][r] * lrow[r]);
    }
}

// ---------------------------------------------------------------------------
// Kernel 3: output projection GEMM (M=8192, N=1024, K=1024), f32 result.
// ---------------------------------------------------------------------------
__global__ void __launch_bounds__(256)
out_gemm_kernel(const __bf16* __restrict__ A, const __bf16* __restrict__ Bt,
                float* __restrict__ Out) {
  __shared__ __align__(16) __bf16 Abuf[2][128 * LDT];
  __shared__ __align__(16) __bf16 Bbuf[2][128 * LDT];
  const int tid = threadIdx.x;
  const int lane = tid & 31, wave = tid >> 5;
  const int wm = wave >> 2, wn = wave & 3;
  const int m0 = blockIdx.y * 128, n0 = blockIdx.x * 128;
  const __bf16* Ag = A + m0 * CD;
  const __bf16* Bg = Bt + n0 * CD;

  auto stage = [&](int buf, int k0) {
    for (int c = tid; c < 512; c += 256) {
      const int row = c >> 2, col = (c & 3) * 8;
      cp16(&Abuf[buf][row * LDT + col], Ag + row * CD + k0 + col);
      cp16(&Bbuf[buf][row * LDT + col], Bg + row * CD + k0 + col);
    }
  };

  v8f acc[4][2] = {};
  stage(0, 0);
  for (int k0 = 0; k0 < CD; k0 += 32) {
    const int cur = (k0 >> 5) & 1;
    if (k0 + 32 < CD) { stage(cur ^ 1, k0 + 32); cp_wait<4>(); }
    else              { cp_wait<0>(); }
    __syncthreads();
    v16bf a[4], bb[2];
#pragma unroll
    for (int i = 0; i < 4; ++i)
      a[i] = load_fragA(&Abuf[cur][(wm * 64 + i * 16) * LDT], LDT, lane);
#pragma unroll
    for (int j = 0; j < 2; ++j)
      bb[j] = load_fragB(&Bbuf[cur][(wn * 32 + j * 16) * LDT], LDT, lane);
#pragma unroll
    for (int i = 0; i < 4; ++i)
#pragma unroll
      for (int j = 0; j < 2; ++j)
        acc[i][j] = wmma_bf16(a[i], bb[j], acc[i][j]);
    __syncthreads();
  }

  const int half = lane >> 4, ln = lane & 15;
  const int mbase = m0 + wm * 64, nbase = n0 + wn * 32;
#pragma unroll
  for (int i = 0; i < 4; ++i)
#pragma unroll
    for (int r = 0; r < 8; ++r) {
      const int row = mbase + i * 16 + r + 8 * half;
#pragma unroll
      for (int j = 0; j < 2; ++j)
        Out[row * CD + nbase + j * 16 + ln] = acc[i][j][r];
    }
}

// ---------------------------------------------------------------------------
extern "C" void kernel_launch(void* const* d_in, const int* in_sizes, int n_in,
                              void* d_out, int out_size, void* d_ws, size_t ws_size,
                              hipStream_t stream) {
  (void)in_sizes; (void)n_in; (void)out_size; (void)ws_size;
  const float* X    = (const float*)d_in[0];   // [4,2048,1024]
  const float* Wqkv = (const float*)d_in[1];   // [1024,3072]
  const float* Wout = (const float*)d_in[2];   // [1024,1024]
  float* out = (float*)d_out;                  // [4,2048,1024]

  char* ws = (char*)d_ws;
  size_t off = 0;
  __bf16* Xbf   = (__bf16*)(ws + off); off += (size_t)CB * CS * CD * 2;
  __bf16* Wqkvt = (__bf16*)(ws + off); off += (size_t)3 * CD * CD * 2;
  __bf16* Woutt = (__bf16*)(ws + off); off += (size_t)CD * CD * 2;
  __bf16* Qw    = (__bf16*)(ws + off); off += (size_t)CB * CS * CD * 2;
  __bf16* Kw    = (__bf16*)(ws + off); off += (size_t)CB * CS * CD * 2;
  __bf16* VTw   = (__bf16*)(ws + off); off += (size_t)CB * CS * CD * 2;
  __bf16* Ctx   = (__bf16*)(ws + off); off += (size_t)CB * CS * CD * 2;

  convert_kernel<<<(CB * CS * CD) / 256, 256, 0, stream>>>(X, Wqkv, Wout,
                                                           Xbf, Wqkvt, Woutt);
  qkv_gemm_kernel<<<dim3(3 * CD / 128, CB * CS / 128), 256, 0, stream>>>(
      Xbf, Wqkvt, Qw, Kw, VTw);
  attn_kernel<<<dim3(CS / 128, CB * CH), 256, 0, stream>>>(Qw, Kw, VTw, Ctx);
  out_gemm_kernel<<<dim3(CD / 128, CB * CS / 128), 256, 0, stream>>>(Ctx, Woutt, out);
}
